// acLSTM_35811437314837
// MI455X (gfx1250) — matrix operand order repaired and checked by reading, not compiled, not run on hardware
//
#include <hip/hip_runtime.h>
#include <hip/hip_bf16.h>
#include <stdint.h>

#define HID   1024
#define INF   171
#define OUTF  171
#define BATCH 128
#define TSEQ  128
#define KP1   1216   // 171 + 1024 padded up to 38*32
#define KP23  2048   // 1024 + 1024
#define NPDEC 176    // 171 padded to 11*16

typedef __attribute__((ext_vector_type(16))) __bf16          v16bf;
typedef __attribute__((ext_vector_type(8)))  float           v8f;
typedef __attribute__((ext_vector_type(8)))  unsigned short  u16x8;

struct U16x16 { u16x8 lo, hi; };

__device__ __forceinline__ v16bf mkfrag(u16x8 lo, u16x8 hi) {
  U16x16 s; s.lo = lo; s.hi = hi;
  return __builtin_bit_cast(v16bf, s);
}

__device__ __forceinline__ unsigned short f2bf(float f) {
  union { float f; unsigned u; } v; v.f = f;
  unsigned r = v.u + 0x7FFFu + ((v.u >> 16) & 1u);   // round-to-nearest-even
  return (unsigned short)(r >> 16);
}

__device__ __forceinline__ float sigm(float x)  { return 1.0f / (1.0f + __expf(-x)); }
__device__ __forceinline__ float tanhx(float x) { return 2.0f / (1.0f + __expf(-2.0f * x)) - 1.0f; }

// CDNA5 async global->LDS copy (16 bytes), tracked by ASYNCcnt.
__device__ __forceinline__ void async_b128_to_lds(unsigned lds_off,
                                                  const unsigned short* src) {
  asm volatile("global_load_async_to_lds_b128 %0, %1, off"
               :: "v"(lds_off), "v"(src)
               : "memory");
}

__device__ __forceinline__ void wait_async0() {
#if __has_builtin(__builtin_amdgcn_s_wait_asynccnt)
  __builtin_amdgcn_s_wait_asynccnt(0);
#else
  asm volatile("s_wait_asynccnt 0x0" ::: "memory");
#endif
}

// ---------------------------------------------------------------------------
// Fused LSTM layer: gates = A(bf16, BATCH x Kpad) @ Wcat^T (bf16, 4H x Kpad),
// then cell update, writing h (bf16) to two destination activation buffers.
// Block: 128 threads = 4 waves; wave w handles M rows [by*64+16w, +16).
// grid.x = HID/16 (j-tile), grid.y = 2 (M halves).
// Each wave computes 4 accumulator tiles: same 16-column j-range in each of
// the i/f/g/o gate quadrants -> lane owns 8 full cell updates at epilogue.
// B tiles are double-buffered in LDS via async global->LDS DMA (ASYNCcnt).
// ---------------------------------------------------------------------------
__global__ __launch_bounds__(128)
void lstm_layer_kernel(const unsigned short* __restrict__ A, int Kpad,
                       const unsigned short* __restrict__ W,     // [4H x Kpad] bf16
                       const float* __restrict__ bsum,           // [4H] (b_ih+b_hh)
                       float* __restrict__ C,                    // [BATCH x HID] cell state
                       unsigned short* __restrict__ H1, int off1, int st1,
                       unsigned short* __restrict__ H2, int off2, int st2)
{
  // 2 buffers x (4 gates x 16 cols x 32 k) bf16 = 2 x 4 KB
  __shared__ __align__(16) unsigned short ldsB[2][64 * 32];

  const int tid  = threadIdx.x;
  const int lane = tid & 31;
  const int wv   = tid >> 5;        // 0..3
  const int half = lane >> 4;       // 0/1
  const int l15  = lane & 15;
  const int j0   = blockIdx.x * 16;
  const int mrow = blockIdx.y * 64 + wv * 16;

  // Per-thread staging geometry: thread covers segments tid and tid+128 of
  // the 256 x 16B tile (row r = s/4 maps to W row q*HID + j0 + (r%16)).
  const int r0 = tid >> 2,          seg0 = tid & 3;
  const int r1 = (tid + 128) >> 2,  seg1 = tid & 3;   // (tid+128)&3 == tid&3
  const int q0_ = r0 >> 4, jj0 = r0 & 15;
  const int q1_ = r1 >> 4, jj1 = r1 & 15;
  const unsigned short* wsrc0 = W + (size_t)(q0_ * HID + j0 + jj0) * Kpad + seg0 * 8;
  const unsigned short* wsrc1 = W + (size_t)(q1_ * HID + j0 + jj1) * Kpad + seg1 * 8;
  const unsigned ldsbase = (unsigned)(uintptr_t)(&ldsB[0][0]);
  const unsigned ldst0 = ldsbase + (unsigned)(r0 * 32 + seg0 * 8) * 2u;
  const unsigned ldst1 = ldsbase + (unsigned)(r1 * 32 + seg1 * 8) * 2u;

  v8f acc[4] = {v8f{}, v8f{}, v8f{}, v8f{}};

  const unsigned short* Arow = A + (size_t)(mrow + l15) * Kpad + half * 8;

  // Prologue: stage first B tile into buffer 0.
  async_b128_to_lds(ldst0, wsrc0);
  async_b128_to_lds(ldst1, wsrc1);
  wait_async0();
  __syncthreads();

  int buf = 0;
  for (int kk = 0; kk < Kpad; kk += 32) {
    // Issue async staging of next tile into the other buffer (no wait yet).
    if (kk + 32 < Kpad) {
      const int nb = buf ^ 1;
      async_b128_to_lds(ldst0 + (unsigned)(nb * 64 * 32 * 2), wsrc0 + kk + 32);
      async_b128_to_lds(ldst1 + (unsigned)(nb * 64 * 32 * 2), wsrc1 + kk + 32);
    }

    // A fragment (16-bit A-matrix layout: lanes<16 K{0..7,16..23},
    // lanes>=16 K{8..15,24..31})
    u16x8 a0 = *(const u16x8*)(Arow + kk);
    u16x8 a1 = *(const u16x8*)(Arow + kk + 16);
    v16bf av = mkfrag(a0, a1);

    #pragma unroll
    for (int q = 0; q < 4; ++q) {
      const unsigned short* bp = &ldsB[buf][(q * 16 + l15) * 32 + half * 8];
      u16x8 b0 = *(const u16x8*)(bp);
      u16x8 b1 = *(const u16x8*)(bp + 16);
      v16bf bv = mkfrag(b0, b1);
      acc[q] = __builtin_amdgcn_wmma_f32_16x16x32_bf16(
          false, av, false, bv, (short)0, acc[q], false, false);
    }

    // My async writes for the next tile are done; everyone finished reading
    // the current buffer once the barrier completes.
    wait_async0();
    __syncthreads();
    buf ^= 1;
  }

  // Epilogue: each lane owns (m = mrow + half*8 + v, j = j0 + l15) for v=0..7
  const int j  = j0 + l15;
  const int mb = mrow + half * 8;
  const float bi = bsum[0 * HID + j];
  const float bf = bsum[1 * HID + j];
  const float bg = bsum[2 * HID + j];
  const float bo = bsum[3 * HID + j];
  #pragma unroll
  for (int v = 0; v < 8; ++v) {
    const int m = mb + v;
    float ig = sigm(acc[0][v] + bi);
    float fg = sigm(acc[1][v] + bf);
    float gt = tanhx(acc[2][v] + bg);
    float og = sigm(acc[3][v] + bo);
    size_t ci = (size_t)m * HID + j;
    float cn = fg * C[ci] + ig * gt;
    float hn = og * tanhx(cn);
    C[ci] = cn;
    unsigned short hb = f2bf(hn);
    H1[(size_t)m * st1 + off1 + j] = hb;
    H2[(size_t)m * st2 + off2 + j] = hb;
  }
}

// ---------------------------------------------------------------------------
// Decoder: out[:, t, :] = h2 @ W_dec^T + b_dec   (128 x 1024 -> 128 x 171)
// ---------------------------------------------------------------------------
__global__ __launch_bounds__(128)
void decoder_kernel(const unsigned short* __restrict__ Ad,     // [BATCH x HID] bf16
                    const unsigned short* __restrict__ Wd,     // [NPDEC x HID] bf16
                    const float* __restrict__ bd,              // [NPDEC]
                    float* __restrict__ out, int t)
{
  const int tid  = threadIdx.x;
  const int lane = tid & 31;
  const int wv   = tid >> 5;
  const int half = lane >> 4;
  const int l15  = lane & 15;
  const int n0   = blockIdx.x * 16;
  const int mrow = blockIdx.y * 64 + wv * 16;

  v8f acc = {};
  const unsigned short* Arow = Ad + (size_t)(mrow + l15) * HID + half * 8;
  const unsigned short* Brow = Wd + (size_t)(n0  + l15) * HID + half * 8;

  for (int kk = 0; kk < HID; kk += 32) {
    u16x8 a0 = *(const u16x8*)(Arow + kk);
    u16x8 a1 = *(const u16x8*)(Arow + kk + 16);
    u16x8 b0 = *(const u16x8*)(Brow + kk);
    u16x8 b1 = *(const u16x8*)(Brow + kk + 16);
    acc = __builtin_amdgcn_wmma_f32_16x16x32_bf16(
        false, mkfrag(a0, a1), false, mkfrag(b0, b1), (short)0, acc, false, false);
  }

  const int n = n0 + l15;
  if (n < OUTF) {
    const float bb = bd[n];
    const int mb = mrow + half * 8;
    #pragma unroll
    for (int v = 0; v < 8; ++v) {
      out[((size_t)(mb + v) * TSEQ + t) * OUTF + n] = acc[v] + bb;
    }
  }
}

// ---------------------------------------------------------------------------
// Per-step input selector: in_frame = cond ? real_seq[:,t,:] : out[:,t-1,:]
// Writes bf16 into the x-region of A1[cur].
// ---------------------------------------------------------------------------
__global__ void prep_input_kernel(const float* __restrict__ seq,
                                  const float* __restrict__ out,
                                  unsigned short* __restrict__ A1, int t,
                                  const int* __restrict__ cn,
                                  const int* __restrict__ gn)
{
  int idx = blockIdx.x * blockDim.x + threadIdx.x;
  if (idx >= BATCH * INF) return;
  int b = idx / INF, col = idx % INF;
  int c = cn[0], g = gn[0];
  int flag = ((t % (c + g)) < g) ? 1 : 0;
  float x;
  if (flag) x = seq[((size_t)b * TSEQ + t) * INF + col];
  else      x = (t > 0) ? out[((size_t)b * TSEQ + (t - 1)) * OUTF + col] : 0.0f;
  A1[(size_t)b * KP1 + col] = f2bf(x);
}

// ---------------------------------------------------------------------------
// Weight pre-conversion (fp32 -> concatenated, K-padded bf16)
// ---------------------------------------------------------------------------
__global__ void conv_w1(const float* __restrict__ Wih, const float* __restrict__ Whh,
                        unsigned short* __restrict__ Wc)
{
  size_t idx = (size_t)blockIdx.x * blockDim.x + threadIdx.x;
  if (idx >= (size_t)4 * HID * KP1) return;
  int n = (int)(idx / KP1), k = (int)(idx % KP1);
  float v = 0.0f;
  if (k < INF)            v = Wih[(size_t)n * INF + k];
  else if (k < INF + HID) v = Whh[(size_t)n * HID + (k - INF)];
  Wc[idx] = f2bf(v);
}

__global__ void conv_w2(const float* __restrict__ Wih, const float* __restrict__ Whh,
                        unsigned short* __restrict__ Wc)
{
  size_t idx = (size_t)blockIdx.x * blockDim.x + threadIdx.x;
  if (idx >= (size_t)4 * HID * KP23) return;
  int n = (int)(idx / KP23), k = (int)(idx % KP23);
  float v = (k < HID) ? Wih[(size_t)n * HID + k] : Whh[(size_t)n * HID + (k - HID)];
  Wc[idx] = f2bf(v);
}

__global__ void conv_wd(const float* __restrict__ W, unsigned short* __restrict__ Wc)
{
  size_t idx = (size_t)blockIdx.x * blockDim.x + threadIdx.x;
  if (idx >= (size_t)NPDEC * HID) return;
  int n = (int)(idx / HID), k = (int)(idx % HID);
  float v = (n < OUTF) ? W[(size_t)n * HID + k] : 0.0f;
  Wc[idx] = f2bf(v);
}

__global__ void bias_kernel(const float* bi1, const float* bh1,
                            const float* bi2, const float* bh2,
                            const float* bi3, const float* bh3,
                            const float* bd,
                            float* bs1, float* bs2, float* bs3, float* bsd)
{
  int i = blockIdx.x * blockDim.x + threadIdx.x;
  if (i < 4 * HID) {
    bs1[i] = bi1[i] + bh1[i];
    bs2[i] = bi2[i] + bh2[i];
    bs3[i] = bi3[i] + bh3[i];
  }
  if (i < NPDEC) bsd[i] = (i < OUTF) ? bd[i] : 0.0f;
}

// ---------------------------------------------------------------------------
extern "C" void kernel_launch(void* const* d_in, const int* in_sizes, int n_in,
                              void* d_out, int out_size, void* d_ws, size_t ws_size,
                              hipStream_t stream)
{
  (void)in_sizes; (void)n_in; (void)out_size; (void)ws_size;

  const float* seq  = (const float*)d_in[0];
  const float* Wih1 = (const float*)d_in[1];
  const float* Whh1 = (const float*)d_in[2];
  const float* bih1 = (const float*)d_in[3];
  const float* bhh1 = (const float*)d_in[4];
  const float* Wih2 = (const float*)d_in[5];
  const float* Whh2 = (const float*)d_in[6];
  const float* bih2 = (const float*)d_in[7];
  const float* bhh2 = (const float*)d_in[8];
  const float* Wih3 = (const float*)d_in[9];
  const float* Whh3 = (const float*)d_in[10];
  const float* bih3 = (const float*)d_in[11];
  const float* bhh3 = (const float*)d_in[12];
  const float* Wdec = (const float*)d_in[13];
  const float* bdec = (const float*)d_in[14];
  const int*   cnum = (const int*)d_in[15];
  const int*   gnum = (const int*)d_in[16];
  float* out = (float*)d_out;

  // Workspace carve-up (256B aligned)
  char* p = (char*)d_ws;
  auto alloc = [&](size_t bytes) -> void* {
    void* r = (void*)p;
    p += (bytes + 255) & ~(size_t)255;
    return r;
  };
  unsigned short* Wc1 = (unsigned short*)alloc((size_t)4 * HID * KP1  * 2);
  unsigned short* Wc2 = (unsigned short*)alloc((size_t)4 * HID * KP23 * 2);
  unsigned short* Wc3 = (unsigned short*)alloc((size_t)4 * HID * KP23 * 2);
  unsigned short* Wcd = (unsigned short*)alloc((size_t)NPDEC * HID * 2);
  float* bs1 = (float*)alloc(4 * HID * 4);
  float* bs2 = (float*)alloc(4 * HID * 4);
  float* bs3 = (float*)alloc(4 * HID * 4);
  float* bsd = (float*)alloc(NPDEC * 4);
  unsigned short* A1[2], * A2[2], * A3[2];
  A1[0] = (unsigned short*)alloc((size_t)BATCH * KP1  * 2);
  A1[1] = (unsigned short*)alloc((size_t)BATCH * KP1  * 2);
  A2[0] = (unsigned short*)alloc((size_t)BATCH * KP23 * 2);
  A2[1] = (unsigned short*)alloc((size_t)BATCH * KP23 * 2);
  A3[0] = (unsigned short*)alloc((size_t)BATCH * KP23 * 2);
  A3[1] = (unsigned short*)alloc((size_t)BATCH * KP23 * 2);
  unsigned short* Ad = (unsigned short*)alloc((size_t)BATCH * HID * 2);
  float* C0 = (float*)alloc((size_t)BATCH * HID * 4);
  float* C1 = (float*)alloc((size_t)BATCH * HID * 4);
  float* C2 = (float*)alloc((size_t)BATCH * HID * 4);

  // Reset state each launch (deterministic): h buffers + pads + cell states = 0
  hipMemsetAsync(A1[0], 0, (size_t)BATCH * KP1  * 2, stream);
  hipMemsetAsync(A1[1], 0, (size_t)BATCH * KP1  * 2, stream);
  hipMemsetAsync(A2[0], 0, (size_t)BATCH * KP23 * 2, stream);
  hipMemsetAsync(A2[1], 0, (size_t)BATCH * KP23 * 2, stream);
  hipMemsetAsync(A3[0], 0, (size_t)BATCH * KP23 * 2, stream);
  hipMemsetAsync(A3[1], 0, (size_t)BATCH * KP23 * 2, stream);
  hipMemsetAsync(C0, 0, (size_t)BATCH * HID * 4, stream);
  hipMemsetAsync(C1, 0, (size_t)BATCH * HID * 4, stream);
  hipMemsetAsync(C2, 0, (size_t)BATCH * HID * 4, stream);

  // Weight conversion (bf16, concatenated [W_ih | W_hh], K padded)
  {
    size_t n1 = (size_t)4 * HID * KP1;
    conv_w1<<<(unsigned)((n1 + 255) / 256), 256, 0, stream>>>(Wih1, Whh1, Wc1);
    size_t n2 = (size_t)4 * HID * KP23;
    conv_w2<<<(unsigned)((n2 + 255) / 256), 256, 0, stream>>>(Wih2, Whh2, Wc2);
    conv_w2<<<(unsigned)((n2 + 255) / 256), 256, 0, stream>>>(Wih3, Whh3, Wc3);
    size_t nd = (size_t)NPDEC * HID;
    conv_wd<<<(unsigned)((nd + 255) / 256), 256, 0, stream>>>(Wdec, Wcd);
    bias_kernel<<<(4 * HID + 255) / 256, 256, 0, stream>>>(
        bih1, bhh1, bih2, bhh2, bih3, bhh3, bdec, bs1, bs2, bs3, bsd);
  }

  const dim3 lgrid(HID / 16, BATCH / 64);   // (64, 2)
  const dim3 dgrid(NPDEC / 16, BATCH / 64); // (11, 2)
  const int prep_blocks = (BATCH * INF + 255) / 256;

  for (int t = 0; t < TSEQ; ++t) {
    const int cur = t & 1, nxt = cur ^ 1;

    prep_input_kernel<<<prep_blocks, 256, 0, stream>>>(seq, out, A1[cur], t, cnum, gnum);

    // Layer 1: reads A1[cur] (x|h0), writes h0 -> A1[nxt].h region and A2[cur].lo
    lstm_layer_kernel<<<lgrid, 128, 0, stream>>>(
        A1[cur], KP1, Wc1, bs1, C0,
        A1[nxt], INF, KP1,
        A2[cur], 0,   KP23);

    // Layer 2: reads A2[cur] (h0|h1), writes h1 -> A2[nxt].hi and A3[cur].lo
    lstm_layer_kernel<<<lgrid, 128, 0, stream>>>(
        A2[cur], KP23, Wc2, bs2, C1,
        A2[nxt], HID, KP23,
        A3[cur], 0,   KP23);

    // Layer 3: reads A3[cur] (h1|h2), writes h2 -> A3[nxt].hi and Ad
    lstm_layer_kernel<<<lgrid, 128, 0, stream>>>(
        A3[cur], KP23, Wc3, bs3, C2,
        A3[nxt], HID, KP23,
        Ad,      0,   HID);

    // Decoder: out[:, t, :] = h2 @ Wdec^T + b
    decoder_kernel<<<dgrid, 128, 0, stream>>>(Ad, Wcd, bsd, out, t);
  }
}
